// SelfAttention_44074954391682
// MI455X (gfx1250) — matrix operand order repaired and checked by reading, not compiled, not run on hardware
//
#include <hip/hip_runtime.h>

#define B_  4
#define C_  128
#define N_  4096

typedef __attribute__((ext_vector_type(16))) __bf16        v16bf;
typedef __attribute__((ext_vector_type(8)))  float         v8f;
typedef __attribute__((ext_vector_type(4)))  unsigned int  v4u;

union ABf { v16bf v; v4u q[2]; };

#define L2E 1.4426950408889634f

// ---------------- Kernel 1: fused QKV 1x1-conv projections ----------------
// x:[B,C,N] f32 -> Qp:[B,N,16] bf16 (query rows, pre-scaled by log2(e)),
//                  Kp:[B,N,16] bf16 (key rows), Vc:[B,C,N] bf16 (channel-major)
__global__ __launch_bounds__(256) void qkv_kernel(
    const float* __restrict__ x,
    const float* __restrict__ wq, const float* __restrict__ bq,
    const float* __restrict__ wk, const float* __restrict__ bk,
    const float* __restrict__ wv, const float* __restrict__ bv,
    __bf16* __restrict__ Qp, __bf16* __restrict__ Kp, __bf16* __restrict__ Vc)
{
    __shared__ __align__(16) float xs[32 * 132];   // [n][c], padded rows (132 dwords)
    const int b  = blockIdx.x >> 7;
    const int n0 = (blockIdx.x & 127) << 5;
    const int tid = threadIdx.x;

    for (int i = tid; i < 128 * 32; i += 256) {
        int c = i >> 5, n = i & 31;
        xs[n * 132 + c] = x[((size_t)(b * C_ + c)) * N_ + n0 + n];
    }
    __syncthreads();

    const int g = tid >> 5, n = tid & 31;
    const float4* xr = (const float4*)(xs + n * 132);
    for (int i = 0; i < 20; ++i) {
        int co = g * 20 + i;                       // 0..159: 16 q | 16 k | 128 v
        const float* wrow; float bias;
        if (co < 16)      { wrow = wq + co * 128;        bias = bq[co];      }
        else if (co < 32) { wrow = wk + (co - 16) * 128; bias = bk[co - 16]; }
        else              { wrow = wv + (co - 32) * 128; bias = bv[co - 32]; }
        const float4* w4 = (const float4*)wrow;
        float acc = bias;
        #pragma unroll
        for (int c4 = 0; c4 < 32; ++c4) {
            float4 a = xr[c4], w = w4[c4];
            acc += a.x * w.x + a.y * w.y + a.z * w.z + a.w * w.w;
        }
        int nn = n0 + n;
        if (co < 16)      Qp[((size_t)(b * N_ + nn)) * 16 + co]        = (__bf16)(acc * L2E);
        else if (co < 32) Kp[((size_t)(b * N_ + nn)) * 16 + (co - 16)] = (__bf16)acc;
        else              Vc[((size_t)(b * C_ + (co - 32))) * N_ + nn] = (__bf16)acc;
    }
}

// ---------------- Kernel 2: flash attention + residual ----------------
// 4 waves/block; each wave owns 16 query rows. S^T = K.Q^T so exp(S^T) lands
// directly in the A-matrix layout of the P.V WMMA. Tiles arrive in LDS via
// GLOBAL_LOAD_ASYNC_TO_LDS_B128 (ASYNCcnt) with double buffering.
__global__ __launch_bounds__(128) void attn_kernel(
    const __bf16* __restrict__ Qp, const __bf16* __restrict__ Kp,
    const __bf16* __restrict__ Vc, const float*  __restrict__ x,
    const float*  __restrict__ gamma, float* __restrict__ out)
{
    __shared__ __align__(16) __bf16 ktile[2][32 * 24];    // [key][16ch], 48B rows
    __shared__ __align__(16) __bf16 vtile[2][128 * 40];   // [ch][32key], 80B rows

    const int tid  = threadIdx.x;
    const int b    = blockIdx.x >> 6;
    const int q0wg = (blockIdx.x & 63) << 6;
    const int wave = tid >> 5, lane = tid & 31;
    const int hi   = lane >> 4, ln = lane & 15;
    const int qbase = q0wg + wave * 16;
    const int key = tid >> 1, half = tid & 1;

    // Q as B-operand of S^T (32ch x 16 queries): elements 0..7 = ch 8*hi..8*hi+7 of query ln,
    // elements 8..15 = ch 16..31 (zero pad).
    ABf qb; qb.q[0] = (v4u){0,0,0,0}; qb.q[1] = (v4u){0,0,0,0};
    {
        const v4u* qs = (const v4u*)(Qp + ((size_t)(b * N_) + qbase + ln) * 16);
        qb.q[0] = qs[hi];
    }

    v8f zero8 = {};
    v8f o[8];
    #pragma unroll
    for (int t = 0; t < 8; ++t) o[t] = zero8;
    float mrun = -3.0e38f, lrun = 0.0f;

    // async-DMA one 32-key K/V tile into LDS buffer `buf`
    auto issueTile = [&](int buf, int kb) {
        unsigned vdst = (unsigned)(unsigned long long)((char*)vtile[buf] + tid * 80);
        unsigned long long vsrc =
            (unsigned long long)(Vc + ((size_t)(b * C_ + tid)) * N_ + kb);
        asm volatile(
            "global_load_async_to_lds_b128 %0, %1, off\n\t"
            "global_load_async_to_lds_b128 %0, %1, off offset:16\n\t"
            "global_load_async_to_lds_b128 %0, %1, off offset:32\n\t"
            "global_load_async_to_lds_b128 %0, %1, off offset:48"
            :: "v"(vdst), "v"(vsrc) : "memory");
        if (tid < 64) {
            unsigned kdst = (unsigned)(unsigned long long)((char*)ktile[buf] + key * 48 + half * 16);
            unsigned long long ksrc =
                (unsigned long long)(Kp + ((size_t)(b * N_) + kb + key) * 16 + half * 8);
            asm volatile("global_load_async_to_lds_b128 %0, %1, off"
                         :: "v"(kdst), "v"(ksrc) : "memory");
        }
    };

    issueTile(0, 0);
    for (int it = 0; it < N_ / 32; ++it) {
        const int cur = it & 1;
        asm volatile("s_wait_asynccnt 0" ::: "memory");  // own async copies landed
        __syncthreads();                                  // everyone's copies landed
        if (it + 1 < N_ / 32) issueTile(cur ^ 1, (it + 1) * 32);  // overlap w/ compute

        const char* kt = (const char*)ktile[cur];
        const char* vt = (const char*)vtile[cur];

        // ---- S^T tiles: A = K rows (16 keys x 32ch, ch 16..31 zero) ----
        ABf a0, a1;
        a0.q[0] = *(const v4u*)(kt + ln * 48 + hi * 16);
        a0.q[1] = (v4u){0,0,0,0};
        a1.q[0] = *(const v4u*)(kt + (16 + ln) * 48 + hi * 16);
        a1.q[1] = (v4u){0,0,0,0};
        v8f st0 = __builtin_amdgcn_wmma_f32_16x16x32_bf16(false, a0.v, false, qb.v, (short)0, zero8, false, false);
        v8f st1 = __builtin_amdgcn_wmma_f32_16x16x32_bf16(false, a1.v, false, qb.v, (short)0, zero8, false, false);

        // ---- online softmax in exp2 domain (Q pre-scaled by log2 e) ----
        float mloc = -3.0e38f;
        #pragma unroll
        for (int j = 0; j < 8; ++j) mloc = fmaxf(mloc, fmaxf(st0[j], st1[j]));
        mloc = fmaxf(mloc, __shfl_xor(mloc, 16, 32));

        const bool need = __any(mloc > mrun);   // wave-uniform: max changed?
        float mnew = need ? fmaxf(mrun, mloc) : mrun;

        ABf pa; float ssum = 0.0f;
        #pragma unroll
        for (int j = 0; j < 8; ++j) {
            float p = __builtin_amdgcn_exp2f(st0[j] - mnew);
            ssum += p; pa.v[j] = (__bf16)p;
        }
        #pragma unroll
        for (int j = 0; j < 8; ++j) {
            float p = __builtin_amdgcn_exp2f(st1[j] - mnew);
            ssum += p; pa.v[8 + j] = (__bf16)p;
        }
        ssum += __shfl_xor(ssum, 16, 32);

        if (need) {
            float alpha = __builtin_amdgcn_exp2f(mrun - mnew);
            lrun = lrun * alpha + ssum;
            mrun = mnew;
            float fac[8];
            #pragma unroll
            for (int v = 0; v < 8; ++v) fac[v] = __shfl(alpha, v + hi * 8, 16);
            #pragma unroll
            for (int t = 0; t < 8; ++t)
                #pragma unroll
                for (int v = 0; v < 8; ++v) o[t][v] *= fac[v];
        } else {
            lrun += ssum;
        }

        // ---- O[16q x 128ch] += P[16x32] . V[32 x 16ch] per 16-ch tile ----
        #pragma unroll
        for (int t = 0; t < 8; ++t) {
            const char* vb0 = vt + (16 * t + ln) * 80 + hi * 16;
            ABf vb; vb.q[0] = *(const v4u*)vb0; vb.q[1] = *(const v4u*)(vb0 + 32);
            o[t] = __builtin_amdgcn_wmma_f32_16x16x32_bf16(false, pa.v, false, vb.v, (short)0, o[t], false, false);
        }
    }

    // ---- epilogue: out = gamma * O / l + x ----
    float linv = 1.0f / lrun;
    float li[8];
    #pragma unroll
    for (int v = 0; v < 8; ++v) li[v] = __shfl(linv, v + hi * 8, 16);
    const float g = gamma[0];
    #pragma unroll
    for (int t = 0; t < 8; ++t) {
        int ch = 16 * t + ln;
        #pragma unroll
        for (int v = 0; v < 8; ++v) {
            int nrow = qbase + v + hi * 8;
            size_t idx = ((size_t)(b * C_ + ch)) * N_ + nrow;
            out[idx] = g * o[t][v] * li[v] + x[idx];
        }
    }
}

extern "C" void kernel_launch(void* const* d_in, const int* in_sizes, int n_in,
                              void* d_out, int out_size, void* d_ws, size_t ws_size,
                              hipStream_t stream) {
    (void)in_sizes; (void)n_in; (void)out_size; (void)ws_size;
    const float* x  = (const float*)d_in[0];
    const float* wq = (const float*)d_in[1];
    const float* bq = (const float*)d_in[2];
    const float* wk = (const float*)d_in[3];
    const float* bk = (const float*)d_in[4];
    const float* wv = (const float*)d_in[5];
    const float* bv = (const float*)d_in[6];
    const float* gm = (const float*)d_in[7];

    __bf16* Qp = (__bf16*)d_ws;                    // [B,N,16]  512 KB
    __bf16* Kp = Qp + (size_t)B_ * N_ * 16;        // [B,N,16]  512 KB
    __bf16* Vc = Kp + (size_t)B_ * N_ * 16;        // [B,C,N]   4 MB

    qkv_kernel<<<dim3(512), dim3(256), 0, stream>>>(x, wq, bq, wk, bk, wv, bv, Qp, Kp, Vc);
    attn_kernel<<<dim3(256), dim3(128), 0, stream>>>(Qp, Kp, Vc, x, gm, (float*)d_out);
}
